// MultiViewGCN_23089744183512
// MI455X (gfx1250) — matrix operand order repaired and checked by reading, not compiled, not run on hardware
//
#include <hip/hip_runtime.h>
#include <hip/hip_bf16.h>
#include <stdint.h>

typedef __attribute__((ext_vector_type(16))) __bf16 v16bf;
typedef __attribute__((ext_vector_type(8)))  float  v8f;
typedef __attribute__((ext_vector_type(2)))  float  v2f;
typedef __attribute__((ext_vector_type(2)))  __bf16 v2bf;

// pack two f32 -> one u32 of two bf16 (lowers to v_cvt_pk_bf16_f32)
static __device__ __forceinline__ uint32_t packf2(float lo, float hi) {
  v2f f; f.x = lo; f.y = hi;
  v2bf b = __builtin_convertvector(f, v2bf);
  union { v2bf b; uint32_t u; } t; t.b = b;
  return t.u;
}

static __device__ __forceinline__ float gelu_exact(float x) {
  return 0.5f * x * (1.0f + erff(x * 0.70710678118654752440f));
}

// LDS row stride: 16 data u32 + 4 pad -> 80B; c*20 mod 64 partitions banks
// into 16 disjoint 4-bank groups => conflict-free b128 per 16-lane phase.
#define LDS_ROW 20

// C = A @ B (+bias).  bf16 WMMA, f32 accum.  Block 128x128, K-tile 32,
// 8 wave32 waves; each wave 64x32 (8 WMMAs / K-step).
// OUT_MODE: 0 = f32 row-major, 1 = bf16 row-major, 2 = bf16 transposed (C^T)
// If !B_F32, B is given TRANSPOSED (B^T, row = n, ldb = stride between n-rows).
template<bool A_F32, bool B_F32, int OUT_MODE>
__global__ __launch_bounds__(256)
void gemm_wmma_kernel(const void* __restrict__ A_, const void* __restrict__ B_,
                      const float* __restrict__ bias, void* __restrict__ C_,
                      int M, int N, int K, int lda, int ldb, int ldc)
{
  __shared__ __align__(16) uint32_t AsU[128 * LDS_ROW];
  __shared__ __align__(16) uint32_t BsU[128 * LDS_ROW];

  const int tid   = threadIdx.x;
  const int lane  = tid & 31;
  const int wave  = tid >> 5;
  const int wm    = wave >> 2;   // 0..1 -> 64-row slab
  const int wn    = wave & 3;    // 0..3 -> 32-col slab
  const int lhalf = lane & 15;
  const int khalf = lane >> 4;   // 0 or 1

  const int row_base = blockIdx.x * 128;
  const int col_base = blockIdx.y * 128;

  const float*  Af = (const float*)A_;
  const __bf16* Ab = (const __bf16*)A_;
  const float*  Bf = (const float*)B_;
  const __bf16* Bb = (const __bf16*)B_;

  v8f acc[4][2];
  #pragma unroll
  for (int i = 0; i < 4; ++i)
    #pragma unroll
    for (int j = 0; j < 2; ++j)
      #pragma unroll
      for (int e = 0; e < 8; ++e) acc[i][j][e] = 0.0f;

  // prefetch registers (only the paths selected by the template are live)
  float4 aF[4]; uint4 aB[2]; float bF[16]; uint4 bB[2];

  auto preload = [&](int kt) {
    const int k0 = kt << 5;
    if constexpr (A_F32) {
      #pragma unroll
      for (int i = 0; i < 4; ++i) {                 // 1024 float4 units / tile
        int s = tid + (i << 8);
        int r = s >> 3, kq = s & 7;                 // kq: float4 within row
        aF[i] = *(const float4*)(Af + (size_t)(row_base + r) * lda + k0 + (kq << 2));
      }
    } else {
      #pragma unroll
      for (int i = 0; i < 2; ++i) {                 // 512 uint4 units / tile
        int g = tid + (i << 8);
        int r = g >> 2, kq = g & 3;                 // kq: uint4 (8 bf16) within row
        aB[i] = *(const uint4*)(Ab + (size_t)(row_base + r) * lda + k0 + (kq << 3));
      }
    }
    if constexpr (B_F32) {
      #pragma unroll
      for (int i = 0; i < 8; ++i) {                 // per-slot strided f32 pair
        int s = tid + (i << 8);
        int kp = s >> 7, c = s & 127;
        int gc = col_base + c, gk = k0 + (kp << 1);
        bool ok = gc < N;
        bF[2 * i]     = ok ? Bf[(size_t)gk * ldb + gc]       : 0.0f;
        bF[2 * i + 1] = ok ? Bf[(size_t)(gk + 1) * ldb + gc] : 0.0f;
      }
    } else {
      #pragma unroll
      for (int i = 0; i < 2; ++i) {                 // B^T: contiguous along K
        int g = tid + (i << 8);
        int c = g >> 2, kq = g & 3;
        int gc = col_base + c;
        if (gc < N) bB[i] = *(const uint4*)(Bb + (size_t)gc * ldb + k0 + (kq << 3));
        else        bB[i] = make_uint4(0u, 0u, 0u, 0u);
      }
    }
  };

  auto store_tile = [&]() {
    if constexpr (A_F32) {
      #pragma unroll
      for (int i = 0; i < 4; ++i) {
        int s = tid + (i << 8);
        int r = s >> 3, kq = s & 7;
        uint2 u;
        u.x = packf2(aF[i].x, aF[i].y);
        u.y = packf2(aF[i].z, aF[i].w);
        *(uint2*)&AsU[r * LDS_ROW + (kq << 1)] = u;
      }
    } else {
      #pragma unroll
      for (int i = 0; i < 2; ++i) {
        int g = tid + (i << 8);
        int r = g >> 2, kq = g & 3;
        *(uint4*)&AsU[r * LDS_ROW + (kq << 2)] = aB[i];
      }
    }
    if constexpr (B_F32) {
      #pragma unroll
      for (int i = 0; i < 8; ++i) {
        int s = tid + (i << 8);
        int kp = s >> 7, c = s & 127;
        BsU[c * LDS_ROW + kp] = packf2(bF[2 * i], bF[2 * i + 1]);
      }
    } else {
      #pragma unroll
      for (int i = 0; i < 2; ++i) {
        int g = tid + (i << 8);
        int c = g >> 2, kq = g & 3;
        *(uint4*)&BsU[c * LDS_ROW + (kq << 2)] = bB[i];
      }
    }
  };

  const int ksteps = K >> 5;
  preload(0);

  for (int kt = 0; kt < ksteps; ++kt) {
    store_tile();
    __syncthreads();
    if (kt + 1 < ksteps) preload(kt + 1);   // overlap HBM with WMMA below

    // Fragment gathers: contiguous b128 pairs straight into operand VGPRs.
    union FragQ { uint4 q[2]; v16bf v; };
    FragQ fa[4], fb[2];
    #pragma unroll
    for (int mt = 0; mt < 4; ++mt) {
      int r = wm * 64 + mt * 16 + lhalf;
      // A 16x32: j=0..3 -> kp = khalf*4 + j ; j=4..7 -> kp = 8 + khalf*4 + (j-4)
      fa[mt].q[0] = *(const uint4*)&AsU[r * LDS_ROW + (khalf << 2)];
      fa[mt].q[1] = *(const uint4*)&AsU[r * LDS_ROW + 8 + (khalf << 2)];
    }
    #pragma unroll
    for (int nt = 0; nt < 2; ++nt) {
      int c = wn * 32 + nt * 16 + lhalf;
      // B 32x16: j=0..3 -> kp = khalf*4 + j ; j=4..7 -> kp = 8 + khalf*4 + (j-4)
      fb[nt].q[0] = *(const uint4*)&BsU[c * LDS_ROW + (khalf << 2)];
      fb[nt].q[1] = *(const uint4*)&BsU[c * LDS_ROW + 8 + (khalf << 2)];
    }

    #pragma unroll
    for (int mt = 0; mt < 4; ++mt)
      #pragma unroll
      for (int nt = 0; nt < 2; ++nt)
        acc[mt][nt] = __builtin_amdgcn_wmma_f32_16x16x32_bf16(
            false, fa[mt].v, false, fb[nt].v, (short)0, acc[mt][nt], false, false);

    __syncthreads();
  }

  // ---- epilogue (C layout: VGPR j -> M = j + 8*khalf, lane 0..15 -> N) ----
  #pragma unroll
  for (int mt = 0; mt < 4; ++mt) {
    #pragma unroll
    for (int nt = 0; nt < 2; ++nt) {
      int gn = col_base + wn * 32 + nt * 16 + lhalf;
      if (gn < N) {
        float bv = bias ? bias[gn] : 0.0f;
        int gm0 = row_base + wm * 64 + mt * 16 + (khalf << 3);
        if constexpr (OUT_MODE == 2) {
          // bf16 transposed: 8 consecutive M per lane -> one b128 store
          uint4 w;
          w.x = packf2(acc[mt][nt][0] + bv, acc[mt][nt][1] + bv);
          w.y = packf2(acc[mt][nt][2] + bv, acc[mt][nt][3] + bv);
          w.z = packf2(acc[mt][nt][4] + bv, acc[mt][nt][5] + bv);
          w.w = packf2(acc[mt][nt][6] + bv, acc[mt][nt][7] + bv);
          *(uint4*)((__bf16*)C_ + (size_t)gn * ldc + gm0) = w;
        } else {
          #pragma unroll
          for (int j = 0; j < 8; ++j) {
            float v = acc[mt][nt][j] + bv;
            if constexpr (OUT_MODE == 1)
              ((__bf16*)C_)[(size_t)(gm0 + j) * ldc + gn] = (__bf16)v;
            else
              ((float*)C_)[(size_t)(gm0 + j) * ldc + gn] = v;
          }
        }
      }
    }
  }
}

// Column mean / rstd over 8192 rows (biased variance, matches jnp.var).
__global__ __launch_bounds__(256)
void colstats_kernel(const float* __restrict__ O, int ncols,
                     float* __restrict__ mean, float* __restrict__ rstd)
{
  __shared__ float s1[256], s2[256];
  const int t  = threadIdx.x;
  const int c  = blockIdx.x * 64 + (t & 63);
  const int rg = t >> 6;
  float sum = 0.0f, sq = 0.0f;
  for (int r = rg; r < 8192; r += 4) {
    float v = O[(size_t)r * ncols + c];
    sum += v; sq += v * v;
  }
  s1[t] = sum; s2[t] = sq;
  __syncthreads();
  if (rg == 0) {
    float S = s1[t] + s1[t + 64] + s1[t + 128] + s1[t + 192];
    float Q = s2[t] + s2[t + 64] + s2[t + 128] + s2[t + 192];
    float m = S * (1.0f / 8192.0f);
    float var = Q * (1.0f / 8192.0f) - m * m;
    mean[c] = m;
    rstd[c] = rsqrtf(var + 1e-5f);
  }
}

// x = gamma*(o-mean)*rstd + beta; exact GELU; -> bf16 (row-major, ld = ncols)
__global__ __launch_bounds__(256)
void bngelu_kernel(const float* __restrict__ O,
                   const float* __restrict__ mean, const float* __restrict__ rstd,
                   const float* __restrict__ gamma, const float* __restrict__ beta,
                   int viewStride, int ncols, __bf16* __restrict__ out)
{
  int idx = blockIdx.x * blockDim.x + threadIdx.x;
  int total = 8192 * ncols;
  if (idx < total) {
    int c = idx % ncols;
    int h = c & 255, v = c >> 8;
    float g = gamma[v * viewStride + h];
    float b = beta[v * viewStride + h];
    float x = g * (O[idx] - mean[c]) * rstd[c] + b;
    out[idx] = (__bf16)gelu_exact(x);
  }
}

// combined = sum_v softmax(view_weights)[v] * X[:, v*256:(v+1)*256]
__global__ __launch_bounds__(256)
void combine_kernel(const __bf16* __restrict__ X, const float* __restrict__ vw,
                    __bf16* __restrict__ out)
{
  float w0 = vw[0], w1 = vw[1], w2 = vw[2];
  float mx = fmaxf(w0, fmaxf(w1, w2));
  float e0 = expf(w0 - mx), e1 = expf(w1 - mx), e2 = expf(w2 - mx);
  float inv = 1.0f / (e0 + e1 + e2);
  int idx = blockIdx.x * blockDim.x + threadIdx.x;
  if (idx < 8192 * 256) {
    int r = idx >> 8, c = idx & 255;
    const __bf16* row = X + (size_t)r * 768;
    float v = (e0 * (float)row[c] + e1 * (float)row[256 + c] + e2 * (float)row[512 + c]) * inv;
    out[idx] = (__bf16)v;
  }
}

extern "C" void kernel_launch(void* const* d_in, const int* in_sizes, int n_in,
                              void* d_out, int out_size, void* d_ws, size_t ws_size,
                              hipStream_t stream)
{
  const float* view0 = (const float*)d_in[0];
  const float* view1 = (const float*)d_in[1];
  const float* view2 = (const float*)d_in[2];
  const float* adj   = (const float*)d_in[3];
  const float* p0w = (const float*)d_in[4];
  const float* p0b = (const float*)d_in[5];
  const float* p1w = (const float*)d_in[6];
  const float* p1b = (const float*)d_in[7];
  const float* p2w = (const float*)d_in[8];
  const float* p2b = (const float*)d_in[9];
  const float* gw  = (const float*)d_in[10];  // [3,2,256,256]
  const float* gb  = (const float*)d_in[11];  // [3,2,256]
  const float* gg  = (const float*)d_in[12];  // gamma [3,2,256]
  const float* gbe = (const float*)d_in[13];  // beta  [3,2,256]
  const float* vw  = (const float*)d_in[14];  // [3]
  const float* c1w = (const float*)d_in[15];
  const float* c1b = (const float*)d_in[16];
  const float* cg  = (const float*)d_in[17];
  const float* cb  = (const float*)d_in[18];
  const float* c2w = (const float*)d_in[19];
  const float* c2b = (const float*)d_in[20];

  char* p = (char*)d_ws;
  __bf16* Xbf  = (__bf16*)p; p += (size_t)8192 * 768 * 2;  // activations [8192 x 768]
  __bf16* STbf = (__bf16*)p; p += (size_t)8192 * 768 * 2;  // supports, TRANSPOSED [768 x 8192]
  float*  Obuf = (float*)p;  p += (size_t)8192 * 768 * 4;  // f32 GEMM out
  __bf16* comb = (__bf16*)p; p += (size_t)8192 * 256 * 2;
  __bf16* hbf  = (__bf16*)p; p += (size_t)8192 * 256 * 2;
  float* meanb = (float*)p;  p += 768 * 4;
  float* rstdb = (float*)p;  p += 768 * 4;

  dim3 blk(256);

  // 1) per-view input projections -> X[8192 x 768] bf16 (row-major)
  const float* vws[3] = {view0, view1, view2};
  const float* pws[3] = {p0w, p1w, p2w};
  const float* pbs[3] = {p0b, p1b, p2b};
  const int    kds[3] = {512, 256, 128};
  for (int v = 0; v < 3; ++v) {
    gemm_wmma_kernel<true, true, 1><<<dim3(64, 2), blk, 0, stream>>>(
        vws[v], pws[v], pbs[v], Xbf + v * 256,
        8192, 256, kds[v], kds[v], 256, 768);
  }

  // 2) two GCN layers, views fused along the 768-wide N dimension
  for (int l = 0; l < 2; ++l) {
    for (int v = 0; v < 3; ++v) {
      // support GEMM, writes S^T so the big GEMM reads B contiguously along K
      gemm_wmma_kernel<false, true, 2><<<dim3(64, 2), blk, 0, stream>>>(
          Xbf + v * 256, gw + (size_t)(v * 2 + l) * 256 * 256, gb + (v * 2 + l) * 256,
          STbf + (size_t)v * 256 * 8192, 8192, 256, 256, 768, 256, 8192);
    }
    // big fused GEMM: adj(f32, bf16-converted in flight) @ S -> O f32
    gemm_wmma_kernel<true, false, 0><<<dim3(64, 6), blk, 0, stream>>>(
        adj, STbf, nullptr, Obuf, 8192, 768, 8192, 8192, 8192, 768);
    colstats_kernel<<<dim3(12), blk, 0, stream>>>(Obuf, 768, meanb, rstdb);
    bngelu_kernel<<<dim3((8192 * 768 + 255) / 256), blk, 0, stream>>>(
        Obuf, meanb, rstdb, gg + l * 256, gbe + l * 256, /*viewStride=*/512, 768, Xbf);
  }

  // 3) adaptive fusion (softmax weights computed on device)
  combine_kernel<<<dim3((8192 * 256 + 255) / 256), blk, 0, stream>>>(Xbf, vw, comb);

  // 4) classifier: Linear -> BN -> GELU -> Linear
  gemm_wmma_kernel<false, true, 0><<<dim3(64, 2), blk, 0, stream>>>(
      comb, c1w, c1b, Obuf, 8192, 256, 256, 256, 256, 256);
  colstats_kernel<<<dim3(4), blk, 0, stream>>>(Obuf, 256, meanb, rstdb);
  bngelu_kernel<<<dim3((8192 * 256 + 255) / 256), blk, 0, stream>>>(
      Obuf, meanb, rstdb, cg, cb, /*viewStride=*/0, 256, hbf);
  gemm_wmma_kernel<false, true, 0><<<dim3(64, 1), blk, 0, stream>>>(
      hbf, c2w, c2b, d_out, 8192, 16, 256, 256, 16, 16);
}
